// Conv2d_37280316129861
// MI455X (gfx1250) — matrix operand order repaired and checked by reading, not compiled, not run on hardware
//
#include <hip/hip_runtime.h>

// ---------------------------------------------------------------------------
// Implicit-GEMM 3x3 conv for MI455X (gfx1250, wave32, WMMA bf16 -> f32 acc).
//   M = 256 (out ch), N = 32*56*56 = 100352 (spatial), K = 128*9 = 1152.
// K is reordered as k = rs*128 + c so each BK=32 tile has tile-uniform (r,s):
// no per-element index math, border predicate hoisted out of the gather loop.
// Weights are repacked once per launch into d_ws as bf16 in (m, rs*128+c)
// layout -> A tiles are staged with async global->LDS b128 (ASYNCcnt), no
// conversions. B (im2col) is gathered, converted to bf16 once, ds_store'd.
// Block tile 128x256xBK32, 8 waves (2x4), wave tile 64x64 = 16 WMMA tiles.
// ---------------------------------------------------------------------------

typedef __attribute__((ext_vector_type(16))) __bf16 v16bf;
typedef __attribute__((ext_vector_type(8)))  __bf16 bf16x8;
typedef __attribute__((ext_vector_type(2)))  __bf16 bf16x2;
typedef __attribute__((ext_vector_type(8)))  float  v8f;
typedef int v4i __attribute__((vector_size(16)));   // matches async builtin param

#define AS1 __attribute__((address_space(1)))
#define AS3 __attribute__((address_space(3)))

static constexpr int N_IMG = 32, C_IN = 128, HH = 56, WW = 56, K_OUT = 256;
static constexpr int HW    = HH * WW;          // 3136
static constexpr int KDIM  = C_IN * 9;         // 1152
static constexpr int NGEMM = N_IMG * HW;       // 100352
static constexpr int BM = 128, BN = 256, BK = 32;
static constexpr int LSTR = 40;                // padded bf16 row stride (80B, 16B-aligned)

__device__ __forceinline__ void async_copy16(const __bf16* g, __bf16* l) {
#if __has_builtin(__builtin_amdgcn_global_load_async_to_lds_b128)
    __builtin_amdgcn_global_load_async_to_lds_b128(
        (AS1 v4i*)g, (AS3 v4i*)l, 0, 0);
#else
    *(v4i*)l = *(const v4i*)g;     // fallback: sync 16B copy
#endif
}

__device__ __forceinline__ void wait_async0() {
#if __has_builtin(__builtin_amdgcn_s_wait_asynccnt)
    __builtin_amdgcn_s_wait_asynccnt(0);
#else
    asm volatile("s_wait_asynccnt 0" ::: "memory");
#endif
}

// One-time (per launch) weight repack: OIHW f32 -> bf16, K reordered to
// k = rs*128 + c.  w2[m][rs*128 + c] = (bf16) wgt[m][c*9 + rs]
__global__ __launch_bounds__(256)
void repack_weights(const float* __restrict__ wgt, __bf16* __restrict__ w2) {
    int i = blockIdx.x * 256 + threadIdx.x;      // < 256*1152
    int m  = i / KDIM;
    int k  = i - m * KDIM;
    int rs = k >> 7;                             // k / 128
    int c  = k & 127;
    w2[(size_t)m * KDIM + k] = (__bf16)wgt[(size_t)m * KDIM + c * 9 + rs];
}

__global__ __launch_bounds__(256)
void conv3x3_wmma_bf16(const float* __restrict__ x,
                       const __bf16* __restrict__ w2,
                       const float* __restrict__ bias,
                       float* __restrict__ out) {
    __shared__ __bf16 sA[2][BM][LSTR];   // weights tile  [m][k]  (20 KB)
    __shared__ __bf16 sB[2][BN][LSTR];   // im2col tile   [n][k]  (40 KB)

    const int tid  = threadIdx.x;
    const int lane = tid & 31;
    const int wave = tid >> 5;
    const int wm   = wave & 1;           // 2 waves along M
    const int wn   = wave >> 1;          // 4 waves along N
    const int m0   = blockIdx.y * BM;
    const int n0   = blockIdx.x * BN;

    // ---- B staging invariants: one gemm-column per thread ----
    const int ng    = n0 + tid;
    const int n_img = ng / HW;
    const int hw    = ng - n_img * HW;
    const int oh    = hw / WW;
    const int ow    = hw - oh * WW;
    const float* xbase = x + (size_t)n_img * C_IN * HW;

    auto stageA = [&](int buf, int kk) {
        // 128 rows x 32 bf16 (64B/row): 512 x 16B lane-loads = 2 rounds.
#pragma unroll
        for (int i = 0; i < 2; ++i) {
            int flat = i * 256 + tid;
            int m    = flat >> 2;
            int seg  = flat & 3;
            async_copy16(w2 + (size_t)(m0 + m) * KDIM + kk + seg * 8,
                         &sA[buf][m][seg * 8]);
        }
    };

    auto stageB = [&](int buf, int kk) {
        const int rs = kk >> 7;          // tile-uniform (K = rs*128 + c)
        const int cb = kk & 127;
        const int r  = rs / 3;
        const int s  = rs - 3 * r;
        const int hi = oh + r - 1;
        const int wi = ow + s - 1;
        __bf16* dst = &sB[buf][tid][0];
        if ((unsigned)hi < (unsigned)HH && (unsigned)wi < (unsigned)WW) {
            const float* src = xbase + (size_t)cb * HW + hi * WW + wi;
#pragma unroll
            for (int j = 0; j < 16; ++j) {
                bf16x2 p = { (__bf16)src[(2 * j) * HW], (__bf16)src[(2 * j + 1) * HW] };
                *(bf16x2*)(dst + 2 * j) = p;
            }
        } else {
#pragma unroll
            for (int j = 0; j < 16; ++j) {
                bf16x2 z = {};
                *(bf16x2*)(dst + 2 * j) = z;
            }
        }
    };

    v8f acc[4][4];
    {
        v8f z = {};
#pragma unroll
        for (int mi = 0; mi < 4; ++mi)
#pragma unroll
            for (int ni = 0; ni < 4; ++ni) acc[mi][ni] = z;
    }

    // Fragment addressing per ISA layouts (wave32):
    //  A 16x32 bf16: lane<16 -> K {0..7,16..23}; lane>=16 -> K {8..15,24..31}
    //  B 32x16 bf16: lane<16 -> K {0..7,16..23}; lane>=16 -> K {8..15,24..31}; N = lane%16
    const int arow = wm * 64 + (lane & 15);
    const int ak0  = (lane >> 4) * 8;
    const int bcol = wn * 64 + (lane & 15);
    const int bk0  = (lane >> 4) * 8;

    stageA(0, 0);
    stageB(0, 0);
    wait_async0();
    __syncthreads();

    const int KSTEPS = KDIM / BK;        // 36
    for (int t = 0; t < KSTEPS; ++t) {
        const int cur = t & 1;
        if (t + 1 < KSTEPS) {            // prefetch next tile into other buffer
            stageA(cur ^ 1, (t + 1) * BK);
            stageB(cur ^ 1, (t + 1) * BK);
        }

        v16bf afrag[4];
#pragma unroll
        for (int mi = 0; mi < 4; ++mi) {
            const __bf16* rp = &sA[cur][arow + mi * 16][0];
            bf16x8 lo = *(const bf16x8*)(rp + ak0);
            bf16x8 hi = *(const bf16x8*)(rp + ak0 + 16);
#pragma unroll
            for (int e = 0; e < 8; ++e) { afrag[mi][e] = lo[e]; afrag[mi][8 + e] = hi[e]; }
        }
        v16bf bfrag[4];
#pragma unroll
        for (int ni = 0; ni < 4; ++ni) {
            const __bf16* cp = &sB[cur][bcol + ni * 16][bk0];
            bf16x8 lo = *(const bf16x8*)cp;
            bf16x8 hi = *(const bf16x8*)(cp + 16);
#pragma unroll
            for (int e = 0; e < 8; ++e) { bfrag[ni][e] = lo[e]; bfrag[ni][8 + e] = hi[e]; }
        }

#pragma unroll
        for (int mi = 0; mi < 4; ++mi)
#pragma unroll
            for (int ni = 0; ni < 4; ++ni)
                acc[mi][ni] = __builtin_amdgcn_wmma_f32_16x16x32_bf16(
                    false, afrag[mi], false, bfrag[ni],
                    (short)0, acc[mi][ni], false, false);

        if (t + 1 < KSTEPS) {
            wait_async0();
            __syncthreads();
        }
    }

    // ---- Epilogue: C/D layout (VGPR r: M=r / r+8, N=lane%16) -> NCHW ----
    const int rowhalf = (lane >> 4) * 8;
    const int nl      = lane & 15;
#pragma unroll
    for (int ni = 0; ni < 4; ++ni) {
        int ngl = n0 + wn * 64 + ni * 16 + nl;
        int nim = ngl / HW;
        int hwl = ngl - nim * HW;
        float* obase = out + (size_t)nim * K_OUT * HW + hwl;
#pragma unroll
        for (int mi = 0; mi < 4; ++mi) {
#pragma unroll
            for (int r = 0; r < 8; ++r) {
                int ko = m0 + wm * 64 + mi * 16 + rowhalf + r;
                obase[(size_t)ko * HW] = acc[mi][ni][r] + bias[ko];
            }
        }
    }
}

extern "C" void kernel_launch(void* const* d_in, const int* in_sizes, int n_in,
                              void* d_out, int out_size, void* d_ws, size_t ws_size,
                              hipStream_t stream) {
    const float* x    = (const float*)d_in[0];   // 32*128*56*56
    const float* wgt  = (const float*)d_in[1];   // 256*128*3*3
    const float* bias = (const float*)d_in[2];   // 256
    float* out        = (float*)d_out;           // 32*256*56*56
    __bf16* w2        = (__bf16*)d_ws;           // needs 256*1152*2 = 576 KB

    repack_weights<<<(K_OUT * KDIM + 255) / 256, 256, 0, stream>>>(wgt, w2);

    dim3 grid(NGEMM / BN, K_OUT / BM, 1);        // (392, 2)
    conv3x3_wmma_bf16<<<grid, 256, 0, stream>>>(x, w2, bias, out);
}
